// VQ1D_39779987095981
// MI455X (gfx1250) — matrix operand order repaired and checked by reading, not compiled, not run on hardware
//
#include <hip/hip_runtime.h>

typedef __bf16 bf16;
typedef __attribute__((ext_vector_type(16))) __bf16 v16bf;
typedef __attribute__((ext_vector_type(8)))  __bf16 v8bf;
typedef __attribute__((ext_vector_type(8)))  float  v8f;

#define B_SZ   8
#define D_DIM  512
#define L_SZ   1024
#define K_CODES 8192
#define BL     (B_SZ * L_SZ)          // 8192 query rows
#define NT_K   (K_CODES / 16)         // 512 column tiles
#define KSPLIT 4
#define CT_PER_SPLIT (NT_K / KSPLIT)  // 128 column tiles per split
#define TILE_BYTES (16 * D_DIM * 2)   // 16 KB bf16 codebook tile

// ---------------------------------------------------------------------------
// CDNA5 async global->LDS copy (16B per lane), tracked by ASYNCcnt.
// LDS address operand is the low 32 bits of the flat address (ISA 10.2).
// ---------------------------------------------------------------------------
__device__ __forceinline__ void async_copy_b128(const void* gsrc, unsigned lds_off) {
  asm volatile("global_load_async_to_lds_b128 %0, %1, off"
               :: "v"(lds_off), "v"(gsrc)
               : "memory");
}
__device__ __forceinline__ void wait_async0() {
#if __has_builtin(__builtin_amdgcn_s_wait_asynccnt)
  __builtin_amdgcn_s_wait_asynccnt(0);
#else
  asm volatile("s_wait_asynccnt 0" ::: "memory");
#endif
}

// ---------------------------------------------------------------------------
// Kernel 1: transpose z_e (B,D,L) -> xT (B*L, D) row-major, fp32 -> bf16
// ---------------------------------------------------------------------------
__global__ __launch_bounds__(256) void vq_prep_x(const float* __restrict__ z_e,
                                                 bf16* __restrict__ xT) {
  __shared__ float tile[32][33];
  const int b  = blockIdx.z;
  const int d0 = blockIdx.y * 32;
  const int l0 = blockIdx.x * 32;
  const int t  = threadIdx.x;

  const float* src = z_e + ((size_t)b * D_DIM + d0) * L_SZ + l0;
  {
    const int li = t & 31, dq = t >> 5;
#pragma unroll
    for (int i = 0; i < 4; ++i) {
      const int d = dq + i * 8;
      tile[d][li] = src[(size_t)d * L_SZ + li];
    }
  }
  __syncthreads();
  bf16* dst = xT + ((size_t)b * L_SZ + l0) * D_DIM + d0;
  {
    const int di = t & 31, lq = t >> 5;
#pragma unroll
    for (int i = 0; i < 4; ++i) {
      const int l = lq + i * 8;
      dst[(size_t)l * D_DIM + di] = (bf16)tile[di][l];
    }
  }
}

// ---------------------------------------------------------------------------
// Kernel 2: codebook fp32 -> bf16 (same layout) + e2[k] = ||e_k||^2
// ---------------------------------------------------------------------------
__global__ __launch_bounds__(256) void vq_prep_cb(const float* __restrict__ cbf,
                                                  bf16* __restrict__ cbb,
                                                  float* __restrict__ e2) {
  const int k = blockIdx.x;
  const float* src = cbf + (size_t)k * D_DIM;
  bf16*        dst = cbb + (size_t)k * D_DIM;
  float ss = 0.f;
#pragma unroll
  for (int i = threadIdx.x; i < D_DIM; i += 256) {
    const float v = src[i];
    ss += v * v;
    dst[i] = (bf16)v;
  }
  __shared__ float red[256];
  red[threadIdx.x] = ss;
  __syncthreads();
  for (int s = 128; s > 0; s >>= 1) {
    if (threadIdx.x < s) red[threadIdx.x] += red[threadIdx.x + s];
    __syncthreads();
  }
  if (threadIdx.x == 0) e2[k] = red[0];
}

// ---------------------------------------------------------------------------
// Kernel 3: main search. 8 waves/WG, each wave owns a 16-row x tile (A held
// stationary in registers). WG double-buffers 16-code bf16 codebook tiles in
// LDS via ASYNC global->LDS copies (no VGPR staging). Per column tile:
// 16x v_wmma_f32_16x16x32_bf16 over D=512, running argmin in registers.
// ---------------------------------------------------------------------------
__global__ __launch_bounds__(256) void vq_main(const bf16* __restrict__ xT,
                                               const bf16* __restrict__ cb,
                                               const float* __restrict__ e2,
                                               float* __restrict__ pmin,
                                               int* __restrict__ pidx) {
  __shared__ bf16 lds_tile[2][16 * D_DIM];   // 2 x 16 KB

  const int wave = threadIdx.x >> 5;
  const int lane = threadIdx.x & 31;
  const int half = lane >> 4;      // which 16-lane half
  const int ln   = lane & 15;
  const int rowBase = (blockIdx.x * 8 + wave) * 16;
  const int ctBase  = blockIdx.y * CT_PER_SPLIT;

  // --- preload A: 16 fragments of (16 rows x 32 dims) bf16 -----------------
  v16bf a[16];
  {
    const bf16* arow = xT + (size_t)(rowBase + ln) * D_DIM;
#pragma unroll
    for (int c = 0; c < 16; ++c) {
      a[c] = *(const v16bf*)(arow + c * 32 + 16 * half);        // K = c*32+16h ..+15
    }
  }

  float minv[8];
  int   mini[8];
#pragma unroll
  for (int i = 0; i < 8; ++i) { minv[i] = 3.4e38f; mini[i] = 0; }

  // async tile fetch: 256 threads x 4 x b128 = 16 KB, VGPR-free
  const char*    cbBytes = (const char*)cb;
  const unsigned ldsBase = (unsigned)(uintptr_t)(&lds_tile[0][0]);
  const unsigned thrOff  = (unsigned)threadIdx.x * 16u;

  auto issue_tile = [&](int ct, int buf) {
    const char* g = cbBytes + (size_t)ct * TILE_BYTES + thrOff;
    const unsigned l = ldsBase + (unsigned)buf * (unsigned)TILE_BYTES + thrOff;
#pragma unroll
    for (int i = 0; i < 4; ++i)
      async_copy_b128(g + i * 4096, l + i * 4096u);
  };

  issue_tile(ctBase, 0);

  for (int t = 0; t < CT_PER_SPLIT; ++t) {
    const int ct = ctBase + t;
    wait_async0();                 // my tile-t copies have landed in LDS
    __syncthreads();               // ... and everyone else's; buf[(t+1)&1] free
    if (t + 1 < CT_PER_SPLIT) issue_tile(ct + 1, (t + 1) & 1);  // overlap w/ compute

    const float e2n  = e2[ct * 16 + ln];           // column norm for lane's code
    const bf16* brow = &lds_tile[t & 1][ln * D_DIM + 16 * half];

    v8f acc0 = {}; v8f acc1 = {};
#pragma unroll
    for (int c = 0; c < 8; ++c) {
      v16bf b0 = *(const v16bf*)(brow + (2 * c)     * 32);
      v16bf b1 = *(const v16bf*)(brow + (2 * c + 1) * 32);
      acc0 = __builtin_amdgcn_wmma_f32_16x16x32_bf16(false, a[2 * c],     false, b0,
                                                     (short)0, acc0, false, false);
      acc1 = __builtin_amdgcn_wmma_f32_16x16x32_bf16(false, a[2 * c + 1], false, b1,
                                                     (short)0, acc1, false, false);
    }

    const int nglob = ct * 16 + ln;
#pragma unroll
    for (int i = 0; i < 8; ++i) {
      const float s = e2n - 2.0f * (acc0[i] + acc1[i]);   // ||x||^2 const per row
      if (s < minv[i]) { minv[i] = s; mini[i] = nglob; }
    }
  }

  // --- reduce across the 16 lanes of each half (rows stay within a half) ---
#pragma unroll
  for (int i = 0; i < 8; ++i) {
    float v = minv[i];
    int   idx = mini[i];
#pragma unroll
    for (int m = 1; m <= 8; m <<= 1) {
      const float ov = __shfl_xor(v, m, 32);
      const int   oi = __shfl_xor(idx, m, 32);
      if (ov < v || (ov == v && oi < idx)) { v = ov; idx = oi; }
    }
    if (ln == 0) {
      const int r = rowBase + half * 8 + i;        // VGPR i: M=i (half0), M=8+i (half1)
      pmin[(size_t)blockIdx.y * BL + r] = v;
      pidx[(size_t)blockIdx.y * BL + r] = idx;
    }
  }
}

// ---------------------------------------------------------------------------
// Kernel 4: merge K-split partials (first-index tie-break, ascending splits)
// ---------------------------------------------------------------------------
__global__ __launch_bounds__(256) void vq_merge(const float* __restrict__ pmin,
                                                const int* __restrict__ pidx,
                                                int* __restrict__ idx_out,
                                                float* __restrict__ idx_out_f) {
  const int r = blockIdx.x * 256 + threadIdx.x;
  if (r >= BL) return;
  float best = pmin[r];
  int   bi   = pidx[r];
#pragma unroll
  for (int s = 1; s < KSPLIT; ++s) {
    const float v  = pmin[(size_t)s * BL + r];
    const int   i2 = pidx[(size_t)s * BL + r];
    if (v < best || (v == best && i2 < bi)) { best = v; bi = i2; }
  }
  idx_out[r]   = bi;
  idx_out_f[r] = (float)bi;
}

// ---------------------------------------------------------------------------
// Kernel 5: z_q[b,d,l] = codebook[idx[b*L+l], d]  (fp32 passthrough == STE fwd)
// ---------------------------------------------------------------------------
__global__ __launch_bounds__(256) void vq_gather(const int* __restrict__ idx,
                                                 const float* __restrict__ cbf,
                                                 float* __restrict__ zq) {
  const int chunk = blockIdx.x;          // 32 chunks of 256 (b,l) rows
  const int b  = chunk >> 2;
  const int l  = ((chunk & 3) * 256) + threadIdx.x;
  const int k  = idx[b * L_SZ + l];
  const float* crow = cbf + (size_t)k * D_DIM;
  float* dst = zq + (size_t)b * D_DIM * L_SZ + l;
  const int d0 = blockIdx.y * 64;
#pragma unroll 4
  for (int d = d0; d < d0 + 64; ++d)
    dst[(size_t)d * L_SZ] = crow[d];
}

// ---------------------------------------------------------------------------
extern "C" void kernel_launch(void* const* d_in, const int* in_sizes, int n_in,
                              void* d_out, int out_size, void* d_ws, size_t ws_size,
                              hipStream_t stream) {
  const float* z_e = (const float*)d_in[0];   // (B, D, L) fp32
  const float* cbf = (const float*)d_in[1];   // (K, D)    fp32
  float* out = (float*)d_out;                 // [z_q (B*D*L) | indices (B*L)]

  char* ws = (char*)d_ws;
  bf16*  xT   = (bf16*)(ws);                                    //  8 MB
  bf16*  cbb  = (bf16*)(ws + (8u << 20));                       //  8 MB
  float* e2   = (float*)(ws + (16u << 20));                     // 32 KB (64 KB slot)
  float* pmin = (float*)(ws + (16u << 20) + (1u << 16));        // 128 KB
  int*   pidx = (int*)  (ws + (16u << 20) + (1u << 16) + KSPLIT * BL * 4);
  int*   idxb = (int*)  (ws + (16u << 20) + (1u << 16) + 2 * KSPLIT * BL * 4);

  vq_prep_x <<<dim3(L_SZ / 32, D_DIM / 32, B_SZ), 256, 0, stream>>>(z_e, xT);
  vq_prep_cb<<<K_CODES, 256, 0, stream>>>(cbf, cbb, e2);
  vq_main   <<<dim3(BL / 16 / 8, KSPLIT), 256, 0, stream>>>(xT, cbb, e2, pmin, pidx);
  vq_merge  <<<BL / 256, 256, 0, stream>>>(pmin, pidx, idxb,
                                           out + (size_t)B_SZ * D_DIM * L_SZ);
  vq_gather <<<dim3(BL / 256, D_DIM / 64), 256, 0, stream>>>(idxb, cbf, out);
}
